// AttentionExpert_24713241821200
// MI455X (gfx1250) — compile-verified, matrix operand order checked by reading, not run on hardware
//
#include <hip/hip_runtime.h>
#include <hip/hip_bf16.h>

// ---------------------------------------------------------------------------
// MI455X (gfx1250) fused multi-head attention forward.
// B=2, S=2048, D=1024, H=16, HD=64.
//  - All GEMMs on v_wmma_f32_16x16x32_bf16 (f32 accumulate).
//  - Row-major tiles staged by the Tensor Data Mover (tensor_load_to_lds,
//    TENSORcnt), software-pipelined one tile deep so the DMA and the global
//    loads for the next tile run underneath the WMMA stream.
//  - Flash-style attention: the 32x2048x2048 score tensor never touches HBM.
// ---------------------------------------------------------------------------

typedef __attribute__((ext_vector_type(16))) __bf16    v16bf;
typedef __attribute__((ext_vector_type(8)))  float     v8f;
typedef __attribute__((ext_vector_type(4)))  unsigned  v4u;
typedef __attribute__((ext_vector_type(8)))  unsigned  v8u;

union Frag16 { v16bf v; uint4 u[2]; };

#define CB 2
#define CS 2048
#define CD 1024
#define CH 16
#define CHD 64
#define CM (CB*CS)   // 4096

__device__ __forceinline__ unsigned short f2bf(float f) {
    unsigned u = __float_as_uint(f);
    u += 0x7FFF + ((u >> 16) & 1);            // round-to-nearest-even
    return (unsigned short)(u >> 16);
}

// Low 32 bits of a generic LDS pointer == LDS byte offset (ISA 10.2 aperture map).
__device__ __forceinline__ unsigned lds_off(const void* p) {
    return (unsigned)(size_t)p;
}

// --------------------------- Tensor Data Mover ------------------------------
// 2D bf16 tile load: `rows` x `cols` elements from a row-major tensor with row
// stride `stride` (elements) into contiguous LDS.  D# per CDNA5 ISA.
// Issued by one wave; EXEC ignored; completion via TENSORcnt.
__device__ __forceinline__ void tdm_load_2d(const void* gptr, unsigned lds,
                                            unsigned cols, unsigned rows,
                                            unsigned stride,
                                            unsigned tdim0, unsigned tdim1) {
    unsigned long long ga = (unsigned long long)(size_t)gptr;
    v4u g0;
    g0[0] = 1u;                                        // count=1 (valid D#)
    g0[1] = lds;                                       // lds_addr (bytes)
    g0[2] = (unsigned)ga;                              // global_addr[31:0]
    g0[3] = ((unsigned)(ga >> 32) & 0x01FFFFFFu)       // global_addr[56:32]
            | (2u << 30);                              // type=2 ("image")
    v8u g1;
    g1[0] = 1u << 16;                                  // data_size=1 -> 2 bytes
    g1[1] = (tdim0 & 0xFFFFu) << 16;                   // tensor_dim0[15:0]
    g1[2] = (tdim0 >> 16) | ((tdim1 & 0xFFFFu) << 16); // dim0[31:16]|dim1[15:0]
    g1[3] = (tdim1 >> 16) | (cols << 16);              // dim1[31:16]|tile_dim0
    g1[4] = rows;                                      // tile_dim1 (tile_dim2=0)
    g1[5] = stride;                                    // tensor_dim0_stride lo32
    g1[6] = 0u;
    g1[7] = 0u;
    asm volatile("tensor_load_to_lds %0, %1" :: "s"(g0), "s"(g1) : "memory");
}
#define TDM_WAIT() asm volatile("s_wait_tensorcnt 0x0" ::: "memory")

// ------------------------------- fp32 -> bf16 -------------------------------
__global__ __launch_bounds__(256)
void cvt_bf16(const float* __restrict__ src, unsigned short* __restrict__ dst, int n4) {
    int i = blockIdx.x * 256 + threadIdx.x;
    if (i < n4) {
        float4 f = ((const float4*)src)[i];
        unsigned short e[4] = { f2bf(f.x), f2bf(f.y), f2bf(f.z), f2bf(f.w) };
        ((uint2*)dst)[i] = *(const uint2*)e;
    }
}

// ------------------------------- WMMA GEMM ----------------------------------
// C[M,N] = A[M,K](bf16 rm) * B[K,N](bf16 rm) + bias[N]
// Block: 256 threads = 8 waves.  Block tile 256x64, BK=64, double-buffered.
// Wave w -> rows [32w,32w+32), all 64 cols: 16 WMMA per k-iter.
// A tiles via TDM ping-pong; B tiles via regs->LDS (transposed) ping-pong.
// OUT_MODE 0: fp32 row-major.  OUT_MODE 1: bf16 scattered to [B,H,S,HD].
template <int OUT_MODE>
__global__ __launch_bounds__(256)
void gemm_bf16(const unsigned short* __restrict__ A,
               const unsigned short* __restrict__ Bm,
               const float* __restrict__ bias,
               void* __restrict__ Cout,
               int M, int N, int K)
{
    __shared__ __align__(16) unsigned short As[2][256 * 64];  // [row][k]  64 KB
    __shared__ __align__(16) unsigned short Bt[2][64 * 64];   // [n][k]    16 KB

    const int tid  = threadIdx.x;
    const int w    = tid >> 5;
    const int l    = tid & 31;
    const int half = l >> 4;
    const int lcol = l & 15;
    const int m0 = blockIdx.x * 256;
    const int n0 = blockIdx.y * 64;

    auto gloadB = [&](int i, uint2* regs) {
        #pragma unroll
        for (int t = 0; t < 4; ++t) {
            int g  = tid + t * 256;
            int kk = g >> 4, c4 = (g & 15) * 4;
            regs[t] = *(const uint2*)(Bm + (size_t)(i * 64 + kk) * N + n0 + c4);
        }
    };
    auto storeB = [&](int buf, const uint2* regs) {
        #pragma unroll
        for (int t = 0; t < 4; ++t) {
            int g  = tid + t * 256;
            int kk = g >> 4, c4 = (g & 15) * 4;
            const unsigned short* e = (const unsigned short*)&regs[t];
            #pragma unroll
            for (int q = 0; q < 4; ++q) Bt[buf][(c4 + q) * 64 + kk] = e[q];
        }
    };

    const int T = K / 64;
    uint2 bregs[4];

    // ---- prologue: tile 0 staged, tile 1 global loads in flight ----
    if (tid < 32)
        tdm_load_2d(A + (size_t)m0 * K, lds_off(As[0]), 64, 256, K, K, M);
    gloadB(0, bregs);
    storeB(0, bregs);
    if (T > 1) gloadB(1, bregs);

    v8f acc[2][4] = {};

    for (int i = 0; i < T; ++i) {
        const int cur = i & 1, nxt = cur ^ 1;
        TDM_WAIT();                 // tile i TDM landed (nop on waves 1..7)
        __syncthreads();            // tile i B stores flushed; buffers[nxt] free
        if (tid < 32 && i + 1 < T)  // TDM for tile i+1 runs under compute
            tdm_load_2d(A + (size_t)m0 * K + (i + 1) * 64, lds_off(As[nxt]),
                        64, 256, K, K, M);

        #pragma unroll
        for (int ks = 0; ks < 2; ++ks) {
            const int kb = ks * 32 + half * 8; // K runs {0..7,16..23}/{8..15,24..31}
            Frag16 af[2];
            #pragma unroll
            for (int mi = 0; mi < 2; ++mi) {
                const int row = w * 32 + mi * 16 + lcol;
                af[mi].u[0] = *(const uint4*)(As[cur] + row * 64 + kb);
                af[mi].u[1] = *(const uint4*)(As[cur] + row * 64 + kb + 16);
            }
            #pragma unroll
            for (int j = 0; j < 4; ++j) {
                Frag16 bfr;
                const int bcol = j * 16 + lcol;
                bfr.u[0] = *(const uint4*)(Bt[cur] + bcol * 64 + kb);
                bfr.u[1] = *(const uint4*)(Bt[cur] + bcol * 64 + kb + 16);
                acc[0][j] = __builtin_amdgcn_wmma_f32_16x16x32_bf16(
                    false, af[0].v, false, bfr.v, (short)0, acc[0][j], false, false);
                acc[1][j] = __builtin_amdgcn_wmma_f32_16x16x32_bf16(
                    false, af[1].v, false, bfr.v, (short)0, acc[1][j], false, false);
            }
        }

        if (i + 1 < T) storeB(nxt, bregs);   // regs loaded one iter ago
        if (i + 2 < T) gloadB(i + 2, bregs); // latency spans next compute
    }

    // Epilogue. C layout: vgpr v -> row v + 8*half, col = lane%16.
    #pragma unroll
    for (int mi = 0; mi < 2; ++mi) {
        #pragma unroll
        for (int j = 0; j < 4; ++j) {
            const int col = n0 + j * 16 + lcol;
            const float bv = bias ? bias[col] : 0.f;
            #pragma unroll
            for (int v = 0; v < 8; ++v) {
                const int row = m0 + w * 32 + mi * 16 + v + half * 8;
                const float val = acc[mi][j][v] + bv;
                if (OUT_MODE == 0) {
                    ((float*)Cout)[(size_t)row * N + col] = val;
                } else {
                    const int b = row >> 11, s = row & (CS - 1);
                    const int h = col >> 6,  hd = col & 63;
                    ((unsigned short*)Cout)[((((size_t)b * CH + h) * CS + s) << 6) + hd]
                        = f2bf(val);
                }
            }
        }
    }
}

// --------------------------- fused flash attention --------------------------
// Grid: (S/64, B*H). Block: 128 threads = 4 waves; wave w owns 16 q-rows.
// KV tiled by 64, double-buffered: K tiles via TDM ping-pong, V tiles via
// regs->LDS (transposed) ping-pong.
__global__ __launch_bounds__(128)
void attention_kernel(const unsigned short* __restrict__ Q,   // [B,H,S,HD] bf16
                      const unsigned short* __restrict__ Kq,  // [B,H,S,HD] bf16
                      const unsigned short* __restrict__ Vq,  // [B,H,S,HD] bf16
                      unsigned short* __restrict__ Oout)      // [B,S,H*HD] bf16
{
    __shared__ __align__(16) unsigned short Ks[2][64 * 64];   // [kv][hd]  16 KB
    __shared__ __align__(16) unsigned short Vs[2][64 * 64];   // [hd][kv]  16 KB
    __shared__ __align__(16) unsigned short Ps[4][16 * 64];   // per-wave P 8 KB

    const float scale = 0.125f;  // HD^-0.5
    const int tid  = threadIdx.x;
    const int w    = tid >> 5;
    const int l    = tid & 31;
    const int half = l >> 4;
    const int lcol = l & 15;
    const int bh = blockIdx.y;
    const int q0 = blockIdx.x * 64;

    const unsigned short* qbase = Q  + (size_t)bh * CS * CHD;
    const unsigned short* kbase = Kq + (size_t)bh * CS * CHD;
    const unsigned short* vbase = Vq + (size_t)bh * CS * CHD;

    auto gloadV = [&](int i, uint2* regs) {
        #pragma unroll
        for (int t = 0; t < 8; ++t) {
            int g = tid + t * 128;
            int r = g >> 4, c4 = (g & 15) * 4;
            regs[t] = *(const uint2*)(vbase + (size_t)(i * 64 + r) * CHD + c4);
        }
    };
    auto storeV = [&](int buf, const uint2* regs) {
        #pragma unroll
        for (int t = 0; t < 8; ++t) {
            int g = tid + t * 128;
            int r = g >> 4, c4 = (g & 15) * 4;
            const unsigned short* e = (const unsigned short*)&regs[t];
            #pragma unroll
            for (int q = 0; q < 4; ++q) Vs[buf][(c4 + q) * 64 + r] = e[q];
        }
    };

    // Q fragments (A layout), 2 k-steps over HD=64, loaded once.
    Frag16 qf[2];
    const int qrow = q0 + w * 16 + lcol;
    #pragma unroll
    for (int kk = 0; kk < 2; ++kk) {
        const int o1 = kk * 32 + half * 8;
        qf[kk].u[0] = *(const uint4*)(qbase + (size_t)qrow * CHD + o1);
        qf[kk].u[1] = *(const uint4*)(qbase + (size_t)qrow * CHD + o1 + 16);
    }

    const int T = CS / 64;   // 32 kv tiles
    uint2 vregs[8];

    // ---- prologue ----
    if (tid < 32)
        tdm_load_2d(kbase, lds_off(Ks[0]), CHD, 64, CHD, CHD, CS);
    gloadV(0, vregs);
    storeV(0, vregs);
    gloadV(1, vregs);

    float m_i[8], l_i[8];
    #pragma unroll
    for (int v = 0; v < 8; ++v) { m_i[v] = -1e30f; l_i[v] = 0.f; }
    v8f o[4] = {};

    for (int i = 0; i < T; ++i) {
        const int cur = i & 1, nxt = cur ^ 1;
        TDM_WAIT();
        __syncthreads();
        if (tid < 32 && i + 1 < T)
            tdm_load_2d(kbase + (size_t)(i + 1) * 64 * CHD, lds_off(Ks[nxt]),
                        CHD, 64, CHD, CHD, CS);

        // ---- S = Q * K^T  (B-frag col n = kv row, k = hd) ----
        v8f sc[4];
        #pragma unroll
        for (int j = 0; j < 4; ++j) {
            v8f z = {};
            Frag16 kf0, kf1;
            const int n = j * 16 + lcol;
            kf0.u[0] = *(const uint4*)(Ks[cur] + n * 64 + half * 8);
            kf0.u[1] = *(const uint4*)(Ks[cur] + n * 64 + 16 + half * 8);
            kf1.u[0] = *(const uint4*)(Ks[cur] + n * 64 + 32 + half * 8);
            kf1.u[1] = *(const uint4*)(Ks[cur] + n * 64 + 48 + half * 8);
            z = __builtin_amdgcn_wmma_f32_16x16x32_bf16(false, qf[0].v, false, kf0.v, (short)0, z, false, false);
            z = __builtin_amdgcn_wmma_f32_16x16x32_bf16(false, qf[1].v, false, kf1.v, (short)0, z, false, false);
            sc[j] = z * scale;
        }

        // ---- online softmax: each row lives across 16 lanes of a half ----
        #pragma unroll
        for (int v = 0; v < 8; ++v) {
            float mx = fmaxf(fmaxf(sc[0][v], sc[1][v]), fmaxf(sc[2][v], sc[3][v]));
            #pragma unroll
            for (int d = 1; d < 16; d <<= 1) mx = fmaxf(mx, __shfl_xor(mx, d, 32));
            const float m_new = fmaxf(m_i[v], mx);
            const float corr  = __expf(m_i[v] - m_new);
            float rs = 0.f;
            #pragma unroll
            for (int j = 0; j < 4; ++j) {
                const float p = __expf(sc[j][v] - m_new);
                sc[j][v] = p;
                rs += p;
            }
            #pragma unroll
            for (int d = 1; d < 16; d <<= 1) rs += __shfl_xor(rs, d, 32);
            l_i[v] = l_i[v] * corr + rs;
            m_i[v] = m_new;
            #pragma unroll
            for (int j = 0; j < 4; ++j) o[j][v] *= corr;
        }

        // ---- C-layout -> A-layout for P via per-wave LDS tile ----
        #pragma unroll
        for (int j = 0; j < 4; ++j)
            #pragma unroll
            for (int v = 0; v < 8; ++v)
                Ps[w][(v + half * 8) * 64 + j * 16 + lcol] = f2bf(sc[j][v]);

        Frag16 pf[2];
        #pragma unroll
        for (int kk = 0; kk < 2; ++kk) {
            const int o1 = kk * 32 + half * 8;
            pf[kk].u[0] = *(const uint4*)(&Ps[w][lcol * 64 + o1]);
            pf[kk].u[1] = *(const uint4*)(&Ps[w][lcol * 64 + o1 + 16]);
        }

        // ---- O += P * V  (B-frag col n = hd, k = kv: transposed Vs) ----
        #pragma unroll
        for (int j = 0; j < 4; ++j) {
            Frag16 vf0, vf1;
            const int n = j * 16 + lcol;
            vf0.u[0] = *(const uint4*)(Vs[cur] + n * 64 + half * 8);
            vf0.u[1] = *(const uint4*)(Vs[cur] + n * 64 + 16 + half * 8);
            vf1.u[0] = *(const uint4*)(Vs[cur] + n * 64 + 32 + half * 8);
            vf1.u[1] = *(const uint4*)(Vs[cur] + n * 64 + 48 + half * 8);
            o[j] = __builtin_amdgcn_wmma_f32_16x16x32_bf16(false, pf[0].v, false, vf0.v, (short)0, o[j], false, false);
            o[j] = __builtin_amdgcn_wmma_f32_16x16x32_bf16(false, pf[1].v, false, vf1.v, (short)0, o[j], false, false);
        }

        if (i + 1 < T) storeV(nxt, vregs);
        if (i + 2 < T) gloadV(i + 2, vregs);
    }

    // ---- normalize + scatter to [B, S, H*HD] (row-major for final GEMM) ----
    const int b = bh >> 4, h = bh & 15;
    #pragma unroll
    for (int v = 0; v < 8; ++v) {
        const float inv = 1.f / l_i[v];
        const int s = q0 + w * 16 + v + half * 8;
        #pragma unroll
        for (int j = 0; j < 4; ++j) {
            const int hd = j * 16 + lcol;
            Oout[((size_t)(b * CS + s)) * (CH * CHD) + h * CHD + hd] = f2bf(o[j][v] * inv);
        }
    }
}

// ------------------------------- launcher -----------------------------------
extern "C" void kernel_launch(void* const* d_in, const int* in_sizes, int n_in,
                              void* d_out, int out_size, void* d_ws, size_t ws_size,
                              hipStream_t stream) {
    (void)in_sizes; (void)n_in; (void)out_size; (void)ws_size;

    const float* x  = (const float*)d_in[0];
    const float* wq = (const float*)d_in[1];
    const float* bq = (const float*)d_in[2];
    const float* wk = (const float*)d_in[3];
    const float* bk = (const float*)d_in[4];
    const float* wv = (const float*)d_in[5];
    const float* bv = (const float*)d_in[6];
    const float* wo = (const float*)d_in[7];
    const float* bo = (const float*)d_in[8];
    float* out = (float*)d_out;

    char* ws = (char*)d_ws;
    const size_t MB = 1024ull * 1024ull;
    unsigned short* xb   = (unsigned short*)(ws +  0 * MB);  // 4096x1024 bf16 (8 MB)
    unsigned short* wqb  = (unsigned short*)(ws +  8 * MB);  // 1024x1024 bf16 (2 MB)
    unsigned short* wkb  = (unsigned short*)(ws + 10 * MB);
    unsigned short* wvb  = (unsigned short*)(ws + 12 * MB);
    unsigned short* wob  = (unsigned short*)(ws + 14 * MB);
    unsigned short* qb   = (unsigned short*)(ws + 16 * MB);  // [B,H,S,HD] bf16 (8 MB)
    unsigned short* kb   = (unsigned short*)(ws + 24 * MB);
    unsigned short* vb   = (unsigned short*)(ws + 32 * MB);
    unsigned short* attn = (unsigned short*)(ws + 40 * MB);  // [B,S,H*HD] bf16 (8 MB)

    const int nx4 = CM * CD / 4, nw4 = CD * CD / 4;
    cvt_bf16<<<(nx4 + 255) / 256, 256, 0, stream>>>(x,  xb,  nx4);
    cvt_bf16<<<(nw4 + 255) / 256, 256, 0, stream>>>(wq, wqb, nw4);
    cvt_bf16<<<(nw4 + 255) / 256, 256, 0, stream>>>(wk, wkb, nw4);
    cvt_bf16<<<(nw4 + 255) / 256, 256, 0, stream>>>(wv, wvb, nw4);
    cvt_bf16<<<(nw4 + 255) / 256, 256, 0, stream>>>(wo, wob, nw4);

    dim3 gg(CM / 256, CD / 64);
    gemm_bf16<1><<<gg, 256, 0, stream>>>(xb, wqb, bq, qb, CM, CD, CD);
    gemm_bf16<1><<<gg, 256, 0, stream>>>(xb, wkb, bk, kb, CM, CD, CD);
    gemm_bf16<1><<<gg, 256, 0, stream>>>(xb, wvb, bv, vb, CM, CD, CD);

    dim3 ga(CS / 64, CB * CH);
    attention_kernel<<<ga, 128, 0, stream>>>(qb, kb, vb, attn);

    gemm_bf16<0><<<gg, 256, 0, stream>>>(attn, wob, bo, out, CM, CD, CD);
}
